// GRU_1288490188969
// MI455X (gfx1250) — matrix operand order repaired and checked by reading, not compiled, not run on hardware
//
#include <hip/hip_runtime.h>
#include <hip/hip_bf16.h>
#include <stdint.h>

#define VOCAB 32000
#define HIDDEN 1024
#define BATCH 32
#define SEQ 128

typedef __attribute__((ext_vector_type(16))) __bf16 v16bf;
typedef __attribute__((ext_vector_type(8)))  float  v8f;
typedef __attribute__((ext_vector_type(4)))  unsigned int u32x4;
typedef __attribute__((ext_vector_type(8)))  int    i32x8;
typedef __attribute__((ext_vector_type(4)))  int    i32x4;

#if __has_builtin(__builtin_amdgcn_tensor_load_to_lds)
#define HAVE_TDM 1
#else
#define HAVE_TDM 0
#endif

union Frag {
    uint4  u[2];
    v16bf  v;
};

__device__ __forceinline__ unsigned short f2bf(float f) {
    unsigned int u = __float_as_uint(f);
    unsigned int r = (u + 0x7FFFu + ((u >> 16) & 1u)) >> 16;
    return (unsigned short)r;
}

__device__ __forceinline__ float sigmoidf_(float x) {
    return 1.0f / (1.0f + __expf(-x));
}

// ---------------------------------------------------------------------------
// Transpose + f32 -> bf16 convert:  src is K x N (row major), dst is N x K.
// block (32, 8), grid (N/32, K/32)
// ---------------------------------------------------------------------------
__global__ void transpose_to_bf16(const float* __restrict__ src,
                                  unsigned short* __restrict__ dst,
                                  int K, int N) {
    __shared__ float tile[32][33];
    int n0 = blockIdx.x * 32;
    int k0 = blockIdx.y * 32;
    int tx = threadIdx.x;      // 0..31
    int ty = threadIdx.y;      // 0..7
#pragma unroll
    for (int i = 0; i < 4; ++i) {
        int k = k0 + ty + i * 8;
        tile[ty + i * 8][tx] = src[(size_t)k * N + (n0 + tx)];
    }
    __syncthreads();
#pragma unroll
    for (int i = 0; i < 4; ++i) {
        int n = n0 + ty + i * 8;
        dst[(size_t)n * K + (k0 + tx)] = f2bf(tile[tx][ty + i * 8]);
    }
}

__global__ void init_h(const float* __restrict__ state,
                       float* __restrict__ Hf,
                       unsigned short* __restrict__ Hbf) {
    int i = blockIdx.x * blockDim.x + threadIdx.x;
    if (i < BATCH * HIDDEN) {
        float v = state[i];
        Hf[i]  = v;
        Hbf[i] = f2bf(v);
    }
}

// ---------------------------------------------------------------------------
// Fragment loaders, contiguous k order (A and B agree; not AMD's table order).
// A (16x32 bf16): lane l<16: row l, K 0..15 contiguous; lane>=16: row l-16,
//                 K 16..31 contiguous.
// B (32x16 bf16 from transposed weight): lane l: col n=l&15,
//                 K = (l>>4)*16 .. +15 contiguous.
// ---------------------------------------------------------------------------
__device__ __forceinline__ v16bf load_a_frag(const unsigned short* base,
                                             int row, int k0, int lane) {
    const uint4* p = reinterpret_cast<const uint4*>(
        base + (size_t)row * HIDDEN + k0 + ((lane >> 4) << 4));
    Frag f;
    f.u[0] = p[0];   // K chunk +0..7
    f.u[1] = p[1];   // K chunk +8..15
    return f.v;
}

__device__ __forceinline__ v16bf load_b_frag(const unsigned short* wT,
                                             int n, int k0, int lane) {
    const uint4* p = reinterpret_cast<const uint4*>(
        wT + (size_t)n * HIDDEN + k0 + ((lane >> 4) << 4));
    Frag f;
    f.u[0] = p[0];
    f.u[1] = p[1];   // 16 contiguous K values per lane-half
    return f.v;
}

// LDS B slab layout: row stride 2080 B (2048 data + 16 B pad per 1024 B,
// matching TDM pad_interval=256DW / pad_amount=4DW). Keeps ds_load_b128
// 16B-aligned and staggers rows across LDS bank groups.
#define BROW_STRIDE 2080

__device__ __forceinline__ v16bf load_b_frag_lds(const unsigned short* Bs,
                                                 int nlocal, int k0, int lane) {
    int s = 2 * k0 + ((lane >> 4) << 5);           // byte offset within row
    int addr = nlocal * BROW_STRIDE + s + ((s >> 10) << 4);
    const uint4* p = reinterpret_cast<const uint4*>(
        reinterpret_cast<const char*>(Bs) + addr);
    Frag f;
    f.u[0] = p[0];
    f.u[1] = p[1];
    return f.v;
}

// ---------------------------------------------------------------------------
// Step stage 1: r and z gates (h_prev @ W_h{r,z} via WMMA, fused gather+bias+
// sigmoid epilogue). grid 64, block 128 (4 waves: {r,z} x {mtile0,1})
// ---------------------------------------------------------------------------
__global__ void gru_stage1(const int* __restrict__ x, int t,
                           const float* __restrict__ Wxr,
                           const float* __restrict__ Wxz,
                           const float* __restrict__ br,
                           const float* __restrict__ bz,
                           const unsigned short* __restrict__ Hbf,
                           const float* __restrict__ Hf,
                           const unsigned short* __restrict__ WtHR,
                           const unsigned short* __restrict__ WtHZ,
                           unsigned short* __restrict__ RH,
                           float* __restrict__ Z) {
    const int lane  = threadIdx.x & 31;
    const int wave  = threadIdx.x >> 5;
    const int gemm  = wave >> 1;        // 0 = r, 1 = z
    const int mtile = wave & 1;
    const int ntile = blockIdx.x;

    const unsigned short* WT = gemm ? WtHZ : WtHR;
    const int arow = mtile * 16 + (lane & 15);
    const int bcol = ntile * 16 + (lane & 15);

    v8f acc = {};
    for (int k0 = 0; k0 < HIDDEN; k0 += 32) {
        v16bf a = load_a_frag(Hbf, arow, k0, lane);
        v16bf b = load_b_frag(WT, bcol, k0, lane);
        acc = __builtin_amdgcn_wmma_f32_16x16x32_bf16(
            false, a, false, b, (short)0, acc, false, false);
    }

    const int j     = ntile * 16 + (lane & 15);
    const int rbase = (lane >> 4) * 8;
    const float bias = gemm ? bz[j] : br[j];
#pragma unroll
    for (int v = 0; v < 8; ++v) {
        int bb  = mtile * 16 + rbase + v;          // batch index
        int tok = x[bb * SEQ + t];
        const float* Wx = gemm ? Wxz : Wxr;
        float val = acc[v] + Wx[(size_t)tok * HIDDEN + j] + bias;
        float g   = sigmoidf_(val);
        if (gemm == 0) {
            RH[(size_t)bb * HIDDEN + j] = f2bf(g * Hf[(size_t)bb * HIDDEN + j]);
        } else {
            Z[(size_t)bb * HIDDEN + j] = g;
        }
    }
}

// ---------------------------------------------------------------------------
// Step stage 2: candidate + state update. grid 64, block 64 (2 waves)
// ---------------------------------------------------------------------------
__global__ void gru_stage2(const int* __restrict__ x, int t,
                           const float* __restrict__ Wxh,
                           const float* __restrict__ bh,
                           const unsigned short* __restrict__ RH,
                           const unsigned short* __restrict__ WtHH,
                           const float* __restrict__ Z,
                           float* __restrict__ Hf,
                           unsigned short* __restrict__ Hbf,
                           unsigned short* __restrict__ Hall) {
    const int lane  = threadIdx.x & 31;
    const int mtile = threadIdx.x >> 5;
    const int ntile = blockIdx.x;

    const int arow = mtile * 16 + (lane & 15);
    const int bcol = ntile * 16 + (lane & 15);

    v8f acc = {};
    for (int k0 = 0; k0 < HIDDEN; k0 += 32) {
        v16bf a = load_a_frag(RH, arow, k0, lane);
        v16bf b = load_b_frag(WtHH, bcol, k0, lane);
        acc = __builtin_amdgcn_wmma_f32_16x16x32_bf16(
            false, a, false, b, (short)0, acc, false, false);
    }

    const int j     = ntile * 16 + (lane & 15);
    const int rbase = (lane >> 4) * 8;
    const float bias = bh[j];
#pragma unroll
    for (int v = 0; v < 8; ++v) {
        int bb  = mtile * 16 + rbase + v;
        int tok = x[bb * SEQ + t];
        float val  = acc[v] + Wxh[(size_t)tok * HIDDEN + j] + bias;
        float cand = tanhf(val);
        size_t idx = (size_t)bb * HIDDEN + j;
        float z  = Z[idx];
        float hp = Hf[idx];
        float hn = z * hp + (1.0f - z) * cand;
        Hf[idx] = hn;
        unsigned short hb = f2bf(hn);
        Hbf[idx] = hb;
        Hall[(size_t)t * (BATCH * HIDDEN) + idx] = hb;
    }
}

// ---------------------------------------------------------------------------
// Output projection: out(4096 x 32000) = Hall(4096x1024) @ W_ho + b_o.
// Per block: 128 rows x 64 cols. The 64x1024 bf16 B slab (shared by all 8
// waves) is staged into LDS once by the Tensor Data Mover (TDM), then the
// K-loop feeds WMMA from ds_load_b128 (B) + global_load_b128 (A).
// grid (500, 32), block 256 (8 waves), LDS = 64 * 2080 B = 130 KB.
// ---------------------------------------------------------------------------
__global__ void out_gemm(const unsigned short* __restrict__ Hall,
                         const unsigned short* __restrict__ WtHO,
                         const float* __restrict__ bo,
                         float* __restrict__ out) {
    __shared__ unsigned short Bs[64 * BROW_STRIDE / 2];

    const int lane = threadIdx.x & 31;
    const int wave = threadIdx.x >> 5;
    const int mbase = blockIdx.y * 128 + wave * 16;
    const int nbase = blockIdx.x * 64;

#if HAVE_TDM
    if (threadIdx.x < 32) {
        // Tensor DMA descriptor: 2D tile, 64 rows x 1024 cols of 2B elements,
        // contiguous rows (stride 1024), LDS padding 4 DW per 256 DW.
        unsigned long long gaddr =
            (unsigned long long)(uintptr_t)(WtHO + (size_t)nbase * HIDDEN);
        unsigned int ldsoff = (unsigned int)(uintptr_t)(void*)&Bs[0];

        u32x4 g0;
        g0[0] = 1u;                                   // count=1 valid descriptor
        g0[1] = ldsoff;                               // lds_addr
        g0[2] = (unsigned int)(gaddr & 0xFFFFFFFFull);
        g0[3] = (unsigned int)((gaddr >> 32) & 0x1FFFFFFull) | (2u << 30); // type=2

        i32x8 g1;
        g1[0] = (int)((1u << 16)      // data_size = 2 bytes
                    | (1u << 20)      // pad_enable
                    | (7u << 22)      // pad_interval = 256 DWORDs (1024 B)
                    | (3u << 25));    // pad_amount  = 4 DWORDs (16 B)
        g1[1] = (int)(1024u << 16);   // tensor_dim0[15:0] (=1024, K extent)
        g1[2] = (int)(64u << 16);     // tensor_dim0[31:16]=0 | tensor_dim1[15:0]=64
        g1[3] = (int)(1024u << 16);   // tensor_dim1[31:16]=0 | tile_dim0=1024
        g1[4] = 64;                   // tile_dim1=64 | tile_dim2=0
        g1[5] = 1024;                 // tensor_dim0_stride[31:0]
        g1[6] = 0;                    // stride0[47:32] | stride1[15:0]
        g1[7] = 0;                    // tensor_dim1_stride[47:16]

        i32x4 g2 = {0, 0, 0, 0};
        i32x4 g3 = {0, 0, 0, 0};
#if __clang_major__ >= 23
        i32x8 g4 = {0, 0, 0, 0, 0, 0, 0, 0};
        __builtin_amdgcn_tensor_load_to_lds(g0, g1, g2, g3, g4, 0);
#else
        __builtin_amdgcn_tensor_load_to_lds(g0, g1, g2, g3, 0);
#endif
        __builtin_amdgcn_s_wait_tensorcnt(0);
    }
    __syncthreads();
#else
    // Fallback: cooperative staged copy with the same padded layout.
    for (int idx = threadIdx.x; idx < 64 * 128; idx += 256) {
        int row = idx >> 7;            // 0..63
        int c   = idx & 127;           // 16B chunk within row
        int s   = c * 16;
        int addr = row * BROW_STRIDE + s + ((s >> 10) << 4);
        *reinterpret_cast<uint4*>(reinterpret_cast<char*>(Bs) + addr) =
            *reinterpret_cast<const uint4*>(WtHO + (size_t)(nbase + row) * HIDDEN + c * 8);
    }
    __syncthreads();
#endif

    const int arow = mbase + (lane & 15);
    const int nloc = lane & 15;

    v8f acc0 = {}, acc1 = {}, acc2 = {}, acc3 = {};
    for (int k0 = 0; k0 < HIDDEN; k0 += 32) {
        // prefetch next K slab of the streaming A operand
        if (k0 + 32 < HIDDEN) {
            __builtin_prefetch(Hall + (size_t)arow * HIDDEN + k0 + 32, 0, 0);
        }
        v16bf a  = load_a_frag(Hall, arow, k0, lane);
        v16bf b0 = load_b_frag_lds(Bs, nloc,      k0, lane);
        v16bf b1 = load_b_frag_lds(Bs, nloc + 16, k0, lane);
        v16bf b2 = load_b_frag_lds(Bs, nloc + 32, k0, lane);
        v16bf b3 = load_b_frag_lds(Bs, nloc + 48, k0, lane);
        acc0 = __builtin_amdgcn_wmma_f32_16x16x32_bf16(false, a, false, b0, (short)0, acc0, false, false);
        acc1 = __builtin_amdgcn_wmma_f32_16x16x32_bf16(false, a, false, b1, (short)0, acc1, false, false);
        acc2 = __builtin_amdgcn_wmma_f32_16x16x32_bf16(false, a, false, b2, (short)0, acc2, false, false);
        acc3 = __builtin_amdgcn_wmma_f32_16x16x32_bf16(false, a, false, b3, (short)0, acc3, false, false);
    }

    const int rbase = (lane >> 4) * 8;
    v8f accs[4] = {acc0, acc1, acc2, acc3};
#pragma unroll
    for (int jt = 0; jt < 4; ++jt) {
        int n = nbase + jt * 16 + (lane & 15);
        float bias = bo[n];
#pragma unroll
        for (int v = 0; v < 8; ++v) {
            int m = mbase + rbase + v;
            out[(size_t)m * VOCAB + n] = accs[jt][v] + bias;
        }
    }
}

__global__ void copy_final(const float* __restrict__ Hf, float* __restrict__ dst) {
    int i = blockIdx.x * blockDim.x + threadIdx.x;
    if (i < BATCH * HIDDEN) dst[i] = Hf[i];
}

// ---------------------------------------------------------------------------
extern "C" void kernel_launch(void* const* d_in, const int* in_sizes, int n_in,
                              void* d_out, int out_size, void* d_ws, size_t ws_size,
                              hipStream_t stream) {
    const int*   x     = (const int*)d_in[0];
    const float* state = (const float*)d_in[1];
    const float* Wxr   = (const float*)d_in[2];
    const float* Whr   = (const float*)d_in[3];
    const float* br    = (const float*)d_in[4];
    const float* Wxz   = (const float*)d_in[5];
    const float* Whz   = (const float*)d_in[6];
    const float* bz    = (const float*)d_in[7];
    const float* Wxh   = (const float*)d_in[8];
    const float* Whh   = (const float*)d_in[9];
    const float* bh    = (const float*)d_in[10];
    const float* Who   = (const float*)d_in[11];
    const float* bo    = (const float*)d_in[12];
    float* out = (float*)d_out;

    // workspace layout (all offsets 256B aligned)
    char* ws = (char*)d_ws;
    size_t off = 0;
    unsigned short* WtHR = (unsigned short*)(ws + off); off += (size_t)HIDDEN * HIDDEN * 2;
    unsigned short* WtHZ = (unsigned short*)(ws + off); off += (size_t)HIDDEN * HIDDEN * 2;
    unsigned short* WtHH = (unsigned short*)(ws + off); off += (size_t)HIDDEN * HIDDEN * 2;
    unsigned short* WtHO = (unsigned short*)(ws + off); off += (size_t)VOCAB * HIDDEN * 2;
    unsigned short* Hall = (unsigned short*)(ws + off); off += (size_t)SEQ * BATCH * HIDDEN * 2;
    float*          Hf   = (float*)(ws + off);          off += (size_t)BATCH * HIDDEN * 4;
    unsigned short* Hbf  = (unsigned short*)(ws + off); off += (size_t)BATCH * HIDDEN * 2;
    unsigned short* RH   = (unsigned short*)(ws + off); off += (size_t)BATCH * HIDDEN * 2;
    float*          Zbuf = (float*)(ws + off);          off += (size_t)BATCH * HIDDEN * 4;

    dim3 tb(32, 8);
    transpose_to_bf16<<<dim3(HIDDEN / 32, HIDDEN / 32), tb, 0, stream>>>(Whr, WtHR, HIDDEN, HIDDEN);
    transpose_to_bf16<<<dim3(HIDDEN / 32, HIDDEN / 32), tb, 0, stream>>>(Whz, WtHZ, HIDDEN, HIDDEN);
    transpose_to_bf16<<<dim3(HIDDEN / 32, HIDDEN / 32), tb, 0, stream>>>(Whh, WtHH, HIDDEN, HIDDEN);
    transpose_to_bf16<<<dim3(VOCAB / 32, HIDDEN / 32), tb, 0, stream>>>(Who, WtHO, HIDDEN, VOCAB);

    init_h<<<(BATCH * HIDDEN + 255) / 256, 256, 0, stream>>>(state, Hf, Hbf);

    for (int t = 0; t < SEQ; ++t) {
        gru_stage1<<<HIDDEN / 16, 128, 0, stream>>>(x, t, Wxr, Wxz, br, bz,
                                                    Hbf, Hf, WtHR, WtHZ, RH, Zbuf);
        gru_stage2<<<HIDDEN / 16, 64, 0, stream>>>(x, t, Wxh, bh, RH, WtHH,
                                                   Zbuf, Hf, Hbf, Hall);
    }

    out_gemm<<<dim3(VOCAB / 64, (SEQ * BATCH) / 128), 256, 0, stream>>>(Hall, WtHO, bo, out);

    copy_final<<<(BATCH * HIDDEN + 255) / 256, 256, 0, stream>>>(
        Hf, out + (size_t)SEQ * BATCH * VOCAB);
}
